// TernaryTransformer_11845519802575
// MI455X (gfx1250) — compile-verified, matrix operand order checked by reading, not run on hardware
//
#include <hip/hip_runtime.h>
#include <hip/hip_bf16.h>
#include <math.h>
#include <stdint.h>

// ---------------------------------------------------------------------------
// Problem constants (match the JAX reference)
// ---------------------------------------------------------------------------
#define BB   32
#define SS   100
#define DD   1024
#define HH   16
#define LL   2
#define VV   32000
#define DFF  4096
#define MM   (BB * SS)          // 3200 token rows

// GEMM tiling
#define BMg   64                // block tile M
#define BNg   128               // block tile N
#define KSTEP 64                // K per async stage (2 f16 WMMA K-steps)
#define APADH 72                // padded LDS row stride in halves (bank spread)

typedef __attribute__((ext_vector_type(16))) _Float16 v16h;
typedef __attribute__((ext_vector_type(8)))  _Float16 v8h;
typedef __attribute__((ext_vector_type(8)))  float    v8f;

#define ASYNC_WAIT() asm volatile("s_wait_asynccnt 0x0" ::: "memory")

// ---------------------------------------------------------------------------
// Helpers
// ---------------------------------------------------------------------------
__device__ __forceinline__ float ternary1(float w, float invg) {
    // q = clip(round(w/(g+eps)), -1, 1); w_eff = {q==0:+1, q==+1:-1, q==-1:0}
    float q = rintf(w * invg);
    q = fminf(1.0f, fmaxf(-1.0f, q));
    return (q == 0.0f) ? 1.0f : ((q == 1.0f) ? -1.0f : 0.0f);
}

// Per-lane f16 fragment from an LDS panel row (f16 elements).
// 16-bit A/B VGPR layout: lanes 0-15 need K {k..k+7} U {k+16..k+23} with k=0,
// lanes 16-31 the same with k=8. Two contiguous 8-half runs -> 2x ds_load_b128.
__device__ __forceinline__ v16h frag_h(const _Float16* rowp, int k1) {
    const v8h lo = *(const v8h*)(rowp + k1);
    const v8h hi = *(const v8h*)(rowp + k1 + 16);
    v16h v;
#pragma unroll
    for (int i = 0; i < 8; i++) { v[i] = lo[i]; v[8 + i] = hi[i]; }
    return v;
}

// ---------------------------------------------------------------------------
// f32 -> f16 convert (8 elements / thread, b128 stores). n % 8 == 0.
// ---------------------------------------------------------------------------
__global__ __launch_bounds__(256) void cvt_f16_kernel(
    const float* __restrict__ s, _Float16* __restrict__ d, long long n)
{
    const long long i = ((long long)blockIdx.x * 256 + threadIdx.x) * 8;
    if (i >= n) return;
    const float4 a = *(const float4*)(s + i);
    const float4 b = *(const float4*)(s + i + 4);
    v8h o;
    o[0] = (_Float16)a.x; o[1] = (_Float16)a.y; o[2] = (_Float16)a.z; o[3] = (_Float16)a.w;
    o[4] = (_Float16)b.x; o[5] = (_Float16)b.y; o[6] = (_Float16)b.z; o[7] = (_Float16)b.w;
    *(v8h*)(d + i) = o;
}

// f32 -> ternary-quantized f16 ({+1,-1,0} exact in f16). One pass over dec_w.
__global__ __launch_bounds__(256) void cvt_tern_f16_kernel(
    const float* __restrict__ s, _Float16* __restrict__ d, long long n,
    const float* __restrict__ gsum)
{
    const long long i = ((long long)blockIdx.x * 256 + threadIdx.x) * 8;
    if (i >= n) return;
    const float g    = gsum[0] * (1.0f / ((float)VV * (float)DD));
    const float invg = 1.0f / (g + 1e-6f);
    const float4 a = *(const float4*)(s + i);
    const float4 b = *(const float4*)(s + i + 4);
    v8h o;
    o[0] = (_Float16)ternary1(a.x, invg); o[1] = (_Float16)ternary1(a.y, invg);
    o[2] = (_Float16)ternary1(a.z, invg); o[3] = (_Float16)ternary1(a.w, invg);
    o[4] = (_Float16)ternary1(b.x, invg); o[5] = (_Float16)ternary1(b.y, invg);
    o[6] = (_Float16)ternary1(b.z, invg); o[7] = (_Float16)ternary1(b.w, invg);
    *(v8h*)(d + i) = o;
}

// ---------------------------------------------------------------------------
// Async stage of one f16 K panel (A: 64x64, B: 128x64 halves) into LDS.
// 48 copy tasks of 4 rows each (8 lanes x b128 = one 64-half row).
// Tasks round-robin across the 8 waves (6 per wave). APADH%8==0 keeps b128
// alignment for both the DMA writes and the ds_load_b128 fragment reads.
// ---------------------------------------------------------------------------
__device__ __forceinline__ void stage_panels(
    const _Float16* __restrict__ A, const _Float16* __restrict__ W, int K, int kk,
    _Float16 (*AsB)[APADH], _Float16 (*BsB)[APADH],
    int mBase0, int nBase0, int wave, int lane)
{
    const int subrow = lane >> 3;          // 0..3
    const int li     = (lane & 7) * 8;     // half offset within 64-wide row
    for (int t = wave; t < (BMg + BNg) / 4; t += 8) {
        const int row = t * 4 + subrow;    // task rows are uniformly A or B
        unsigned long long gsrc;
        unsigned lds;
        if (row < BMg) {
            gsrc = (unsigned long long)(A + (size_t)(mBase0 + row) * K + kk + li);
            lds  = (unsigned)(uintptr_t)&AsB[row][li];
        } else {
            const int rb = row - BMg;
            gsrc = (unsigned long long)(W + (size_t)(nBase0 + rb) * K + kk + li);
            lds  = (unsigned)(uintptr_t)&BsB[rb][li];
        }
        asm volatile("global_load_async_to_lds_b128 %0, %1, off"
                     :: "v"(lds), "v"(gsrc) : "memory");
    }
}

// ---------------------------------------------------------------------------
// Embedding gather: h[row,:] = emb[x[row],:] + pos_emb[row % S,:]
// ---------------------------------------------------------------------------
__global__ __launch_bounds__(256) void embed_kernel(
    const int* __restrict__ x, const float* __restrict__ emb,
    const float* __restrict__ pos, float* __restrict__ h)
{
    const int row = blockIdx.x;            // [0, MM)
    const int s   = row % SS;
    const int tok = x[row];
    const float* e = emb + (size_t)tok * DD;
    const float* p = pos + (size_t)s * DD;
    float* o = h + (size_t)row * DD;
    for (int c = threadIdx.x; c < DD; c += 256) o[c] = e[c] + p[c];
}

// ---------------------------------------------------------------------------
// WMMA GEMM: C[M,N] = A[M,K] * W[N,K]^T + bias[N]   (f16 in, f32 out)
//   - async global->LDS double-buffered 64-wide K stages (ASYNCcnt)
//   - 8 waves/block, block tile 64x128; wave tile 16x64 -> 8 WMMAs per stage
//   - all 10 fragments of a stage are preloaded into registers before the 8
//     WMMAs issue, so the matrix ops go back-to-back (no per-WMMA dscnt-0
//     stalls) and co-execute with the next stage's async DMA
//   EPI: 0 = bias only, 1 = bias then exact GELU
// Requires M%64==0, N%128==0, K%64==0 (true for every GEMM here), so EXEC is
// all-1s everywhere (WMMA requirement).
// ---------------------------------------------------------------------------
template <int EPI>
__global__ __launch_bounds__(256) void gemm_wmma(
    const _Float16* __restrict__ A, const _Float16* __restrict__ W,
    const float* __restrict__ bias, float* __restrict__ C,
    int M, int N, int K)
{
    __shared__ _Float16 As[2][BMg][APADH];   // 2 x 64 x 72 f16  (18 KB)
    __shared__ _Float16 Bs[2][BNg][APADH];   // 2 x 128 x 72 f16 (36 KB)

    const int lane = threadIdx.x & 31;
    const int wave = threadIdx.x >> 5;
    const int l16  = lane & 15;
    const int half = lane >> 4;
    const int mw   = wave >> 1;              // 0..3 : wave row within block
    const int nw   = wave & 1;               // 0..1 : wave col within block
    const int mBase0 = blockIdx.y * BMg;
    const int nBase0 = blockIdx.x * BNg;

    v8f acc[4] = {{}, {}, {}, {}};
    const int k1 = half * 8;                 // per-lane K base within a 32-substep

    // prologue: stage panel 0
    stage_panels(A, W, K, 0, As[0], Bs[0], mBase0, nBase0, wave, lane);
    ASYNC_WAIT();
    __syncthreads();

    for (int kk = 0; kk < K; kk += KSTEP) {
        const int buf = (kk / KSTEP) & 1;

        // overlap: DMA the next stage before computing this one. Safe: the
        // barrier at the end of the previous iteration guarantees everyone is
        // done reading buf^1.
        if (kk + KSTEP < K)
            stage_panels(A, W, K, kk + KSTEP, As[buf ^ 1], Bs[buf ^ 1],
                         mBase0, nBase0, wave, lane);

        // ---- preload ALL fragments for this stage (20 ds_load_b128) ----
        v16h av[2];
        v16h bv[2][4];
#pragma unroll
        for (int s2 = 0; s2 < 2; s2++) {
            const int ks = s2 * 32;
            av[s2] = frag_h(&As[buf][mw * 16 + l16][ks], k1);
#pragma unroll
            for (int g = 0; g < 4; g++)
                bv[s2][g] = frag_h(&Bs[buf][nw * 64 + g * 16 + l16][ks], k1);
        }

        // ---- 8 back-to-back WMMAs (no RAW hazards between them) ----
#pragma unroll
        for (int s2 = 0; s2 < 2; s2++) {
#pragma unroll
            for (int g = 0; g < 4; g++) {
                acc[g] = __builtin_amdgcn_wmma_f32_16x16x32_f16(
                    /*neg_a=*/false, av[s2], /*neg_b=*/false, bv[s2][g],
                    /*c_mod=*/(short)0, acc[g], /*reuse_a=*/false, /*reuse_b=*/false);
            }
        }

        if (kk + KSTEP < K) {
            ASYNC_WAIT();                    // own async copies landed in LDS
            __syncthreads();                 // everyone's copies landed
        }
    }

    // epilogue: bias (+ optional exact GELU), store 4 16x16 tiles (f32)
    const int nb   = nBase0 + nw * 64;
    const int mrow = mBase0 + mw * 16 + 8 * half;
#pragma unroll
    for (int g = 0; g < 4; g++) {
        const float bval = bias[nb + g * 16 + l16];
        float* cp = C + (size_t)mrow * N + (nb + g * 16 + l16);
#pragma unroll
        for (int r = 0; r < 8; r++) {        // VGPR r -> row mrow + r
            float v = acc[g][r] + bval;
            if (EPI == 1) v = 0.5f * v * (1.0f + erff(v * 0.70710678118654752f));
            cp[(size_t)r * N] = v;
        }
    }
}

// ---------------------------------------------------------------------------
// Attention: one (b,h) per block, 4 waves, one query row per wave.
// qkv layout [B,S,3D]; q at +0, k at +D, v at +2D within the last dim.
// 0.35% of model FLOPs -> VALU dot products are fine.
// ---------------------------------------------------------------------------
__global__ __launch_bounds__(128) void attention_kernel(
    const float* __restrict__ qkv, float* __restrict__ ctx)
{
    const int bh = blockIdx.x;
    const int b  = bh / HH;
    const int hh = bh % HH;
    const float* base = qkv + (size_t)b * SS * (3 * DD);

    __shared__ float srow[4][128];
    const int wave = threadIdx.x >> 5;
    const int lane = threadIdx.x & 31;

    for (int qi = wave; qi < SS; qi += 4) {
        const float4* qp = (const float4*)(base + (size_t)qi * (3 * DD) + hh * 64);
        float4 qreg[16];
#pragma unroll
        for (int i = 0; i < 16; i++) qreg[i] = qp[i];

        float sc[4];
#pragma unroll
        for (int t = 0; t < 4; t++) {
            const int j = lane + t * 32;
            if (j < SS) {
                const float4* kp = (const float4*)(base + (size_t)j * (3 * DD) + DD + hh * 64);
                float a = 0.0f;
#pragma unroll
                for (int i = 0; i < 16; i++) {
                    const float4 kv = kp[i];
                    a += qreg[i].x * kv.x + qreg[i].y * kv.y
                       + qreg[i].z * kv.z + qreg[i].w * kv.w;
                }
                sc[t] = a * 0.125f;          // 1/sqrt(64)
            } else {
                sc[t] = -3.0e38f;
            }
        }

        float mx = fmaxf(fmaxf(sc[0], sc[1]), fmaxf(sc[2], sc[3]));
        for (int m = 16; m >= 1; m >>= 1) mx = fmaxf(mx, __shfl_xor(mx, m, 32));

        float lsum = 0.0f;
#pragma unroll
        for (int t = 0; t < 4; t++) {
            const int j = lane + t * 32;
            const float e = (j < SS) ? expf(sc[t] - mx) : 0.0f;
            srow[wave][j] = e;
            lsum += e;
        }
        for (int m = 16; m >= 1; m >>= 1) lsum += __shfl_xor(lsum, m, 32);
        const float inv = 1.0f / lsum;

        float c0 = 0.0f, c1 = 0.0f;
        for (int j = 0; j < SS; j++) {
            const float a = srow[wave][j];
            const float* vp = base + (size_t)j * (3 * DD) + 2 * DD + hh * 64;
            c0 += a * vp[lane];
            c1 += a * vp[lane + 32];
        }
        float* op = ctx + ((size_t)b * SS + qi) * DD + hh * 64;
        op[lane]      = c0 * inv;
        op[lane + 32] = c1 * inv;
    }
}

// ---------------------------------------------------------------------------
// Fused residual + LayerNorm (post-norm): out = LN(x + add) * scale + bias
// One block per row of D=1024. Safe in-place on x (reads precede writes).
// ---------------------------------------------------------------------------
__global__ __launch_bounds__(256) void ln_residual_kernel(
    const float* __restrict__ x, const float* __restrict__ add,
    const float* __restrict__ scale, const float* __restrict__ bias,
    float* __restrict__ out)
{
    const int row = blockIdx.x;
    const float* xr = x   + (size_t)row * DD;
    const float* ar = add + (size_t)row * DD;

    float v[4];
    float s = 0.0f, s2 = 0.0f;
#pragma unroll
    for (int t = 0; t < 4; t++) {
        const int c = threadIdx.x + t * 256;
        v[t] = xr[c] + ar[c];
        s  += v[t];
        s2 += v[t] * v[t];
    }

    __shared__ float r1[256], r2[256];
    r1[threadIdx.x] = s; r2[threadIdx.x] = s2;
    __syncthreads();
    for (int st = 128; st > 0; st >>= 1) {
        if (threadIdx.x < st) {
            r1[threadIdx.x] += r1[threadIdx.x + st];
            r2[threadIdx.x] += r2[threadIdx.x + st];
        }
        __syncthreads();
    }
    const float mean = r1[0] * (1.0f / (float)DD);
    const float var  = r2[0] * (1.0f / (float)DD) - mean * mean;
    const float rstd = rsqrtf(var + 1e-5f);

    float* orow = out + (size_t)row * DD;
#pragma unroll
    for (int t = 0; t < 4; t++) {
        const int c = threadIdx.x + t * 256;
        orow[c] = (v[t] - mean) * rstd * scale[c] + bias[c];
    }
}

// ---------------------------------------------------------------------------
// Grid-stride |W| sum for the ternary scale g = mean(|dec_w|)
// ---------------------------------------------------------------------------
__global__ __launch_bounds__(256) void abs_sum_kernel(
    const float* __restrict__ w, long long n, float* __restrict__ gsum)
{
    long long i = (long long)blockIdx.x * blockDim.x + threadIdx.x;
    const long long stride = (long long)gridDim.x * blockDim.x;
    float s = 0.0f;
    for (; i < n; i += stride) s += fabsf(w[i]);

    __shared__ float red[256];
    red[threadIdx.x] = s;
    __syncthreads();
    for (int st = 128; st > 0; st >>= 1) {
        if (threadIdx.x < st) red[threadIdx.x] += red[threadIdx.x + st];
        __syncthreads();
    }
    if (threadIdx.x == 0) atomicAdd(gsum, red[0]);
}

// ---------------------------------------------------------------------------
// Host-side orchestration
// ---------------------------------------------------------------------------
static inline unsigned cvt_blocks(long long n) { return (unsigned)((n / 8 + 255) / 256); }

extern "C" void kernel_launch(void* const* d_in, const int* in_sizes, int n_in,
                              void* d_out, int out_size, void* d_ws, size_t ws_size,
                              hipStream_t stream)
{
    const int*   x      = (const int*)  d_in[0];
    const float* emb    = (const float*)d_in[1];
    const float* pos    = (const float*)d_in[2];
    const float* qkv_w  = (const float*)d_in[3];
    const float* qkv_b  = (const float*)d_in[4];
    const float* out_w  = (const float*)d_in[5];
    const float* out_b  = (const float*)d_in[6];
    const float* ln1_s  = (const float*)d_in[7];
    const float* ln1_b  = (const float*)d_in[8];
    const float* ff1_w  = (const float*)d_in[9];
    const float* ff1_b  = (const float*)d_in[10];
    const float* ff2_w  = (const float*)d_in[11];
    const float* ff2_b  = (const float*)d_in[12];
    const float* ln2_s  = (const float*)d_in[13];
    const float* ln2_b  = (const float*)d_in[14];
    const float* dec_w  = (const float*)d_in[15];
    const float* dec_b  = (const float*)d_in[16];
    float* logits = (float*)d_out;

    // workspace carve-up: f32 activations + f16 staging buffers (~210 MB)
    float* h    = (float*)d_ws;                          // [MM, DD]
    float* qkv  = h   + (size_t)MM * DD;                 // [MM, 3D]
    float* ctx  = qkv + (size_t)MM * 3 * DD;             // [MM, DD]
    float* ff   = ctx + (size_t)MM * DD;                 // [MM, DFF]
    float* gsum = ff  + (size_t)MM * DFF;                // scalar
    _Float16* hA = (_Float16*)(gsum + 64);               // f16 A, max MM x DFF
    _Float16* hW = hA + (size_t)MM * DFF;                // f16 W, max VV x DD

    // 1) embeddings
    embed_kernel<<<MM, 256, 0, stream>>>(x, emb, pos, h);

    // 2) transformer layers
    for (int l = 0; l < LL; l++) {
        // qkv = h @ qkv_w[l]^T + qkv_b[l]     [3200, 3072]
        cvt_f16_kernel<<<cvt_blocks((long long)MM * DD), 256, 0, stream>>>(h, hA, (long long)MM * DD);
        cvt_f16_kernel<<<cvt_blocks((long long)3 * DD * DD), 256, 0, stream>>>(
            qkv_w + (size_t)l * 3 * DD * DD, hW, (long long)3 * DD * DD);
        gemm_wmma<0><<<dim3(3 * DD / BNg, MM / BMg), 256, 0, stream>>>(
            hA, hW, qkv_b + (size_t)l * 3 * DD, qkv, MM, 3 * DD, DD);

        // attention -> ctx [3200, 1024]
        attention_kernel<<<BB * HH, 128, 0, stream>>>(qkv, ctx);

        // attn_out = ctx @ out_w[l]^T + out_b[l]   (into ff scratch)
        cvt_f16_kernel<<<cvt_blocks((long long)MM * DD), 256, 0, stream>>>(ctx, hA, (long long)MM * DD);
        cvt_f16_kernel<<<cvt_blocks((long long)DD * DD), 256, 0, stream>>>(
            out_w + (size_t)l * DD * DD, hW, (long long)DD * DD);
        gemm_wmma<0><<<dim3(DD / BNg, MM / BMg), 256, 0, stream>>>(
            hA, hW, out_b + (size_t)l * DD, ff, MM, DD, DD);

        // h = LN(h + attn_out)
        ln_residual_kernel<<<MM, 256, 0, stream>>>(
            h, ff, ln1_s + (size_t)l * DD, ln1_b + (size_t)l * DD, h);

        // ff = gelu(h @ ff1_w[l]^T + ff1_b[l])   [3200, 4096]
        cvt_f16_kernel<<<cvt_blocks((long long)MM * DD), 256, 0, stream>>>(h, hA, (long long)MM * DD);
        cvt_f16_kernel<<<cvt_blocks((long long)DFF * DD), 256, 0, stream>>>(
            ff1_w + (size_t)l * DFF * DD, hW, (long long)DFF * DD);
        gemm_wmma<1><<<dim3(DFF / BNg, MM / BMg), 256, 0, stream>>>(
            hA, hW, ff1_b + (size_t)l * DFF, ff, MM, DFF, DD);

        // ctx = ff @ ff2_w[l]^T + ff2_b[l]   [3200, 1024]
        cvt_f16_kernel<<<cvt_blocks((long long)MM * DFF), 256, 0, stream>>>(ff, hA, (long long)MM * DFF);
        cvt_f16_kernel<<<cvt_blocks((long long)DD * DFF), 256, 0, stream>>>(
            ff2_w + (size_t)l * DD * DFF, hW, (long long)DD * DFF);
        gemm_wmma<0><<<dim3(DD / BNg, MM / BMg), 256, 0, stream>>>(
            hA, hW, ff2_b + (size_t)l * DD, ctx, MM, DD, DFF);

        // h = LN(h + ff_out)
        ln_residual_kernel<<<MM, 256, 0, stream>>>(
            h, ctx, ln2_s + (size_t)l * DD, ln2_b + (size_t)l * DD, h);
    }

    // 3) ternary decoder: g = mean|dec_w|, quantize once to f16, then
    //    logits = h @ w_eff^T + dec_b
    hipMemsetAsync(gsum, 0, sizeof(float), stream);
    abs_sum_kernel<<<2048, 256, 0, stream>>>(dec_w, (long long)VV * DD, gsum);
    cvt_tern_f16_kernel<<<cvt_blocks((long long)VV * DD), 256, 0, stream>>>(
        dec_w, hW, (long long)VV * DD, gsum);
    cvt_f16_kernel<<<cvt_blocks((long long)MM * DD), 256, 0, stream>>>(h, hA, (long long)MM * DD);
    gemm_wmma<0><<<dim3(VV / BNg, MM / BMg), 256, 0, stream>>>(
        hA, hW, dec_b, logits, MM, VV, DD);
}
